// TiedAxialSelfAttention2d_11931419148696
// MI455X (gfx1250) — matrix operand order repaired and checked by reading, not compiled, not run on hardware
//
#include <hip/hip_runtime.h>
#include <hip/hip_bf16.h>

typedef __attribute__((ext_vector_type(16))) _Float16 v16h;
typedef __attribute__((ext_vector_type(8)))  float    v8f;

#define H_  12
#define DH  64
#define D_  768
#define S_  128
#define L_  256
#define N_  (S_*L_)     // 32768 rows
#define D3  (3*D_)      // 2304

// ---------------- WMMA fragment helpers (CDNA5 wave32 layouts) ----------------

// A fragment 16x32 f16, row-major source, lda in elements.
// Lanes 0-15 -> M=lane, K base 0; lanes 16-31 -> M=lane-16, K base +8.
// VGPR pair g: K = 2g (g<4) else 16+2(g-4).
static __device__ __forceinline__ v16h load_a16(const _Float16* __restrict__ A, int lda) {
  int lane = threadIdx.x & 31;
  int m  = lane & 15;
  int kb = (lane >> 4) ? 8 : 0;
  const _Float16* p = A + (size_t)m * lda + kb;
  v16h a;
#pragma unroll
  for (int g = 0; g < 8; ++g) {
    int k = (g < 4) ? (2*g) : (8 + 2*g);   // 0,2,4,6,16,18,20,22
    a[2*g]   = p[k];
    a[2*g+1] = p[k+1];
  }
  return a;
}

// Same but source is f32 (softmax probabilities), converted on load.
static __device__ __forceinline__ v16h load_a16_f32(const float* __restrict__ A, int lda) {
  int lane = threadIdx.x & 31;
  int m  = lane & 15;
  int kb = (lane >> 4) ? 8 : 0;
  const float* p = A + (size_t)m * lda + kb;
  v16h a;
#pragma unroll
  for (int g = 0; g < 8; ++g) {
    int k = (g < 4) ? (2*g) : (8 + 2*g);
    a[2*g]   = (_Float16)p[k];
    a[2*g+1] = (_Float16)p[k+1];
  }
  return a;
}

// B fragment 32x16 f16 where B[k][n] = W[n*ldw + k]  (source stored [n][k]).
// N = lane&15; lanes 0-15 -> K 0-7,16-23, lanes 16-31 -> K 8-15,24-31 (same order as A).
static __device__ __forceinline__ v16h load_b16_t(const _Float16* __restrict__ W, int ldw) {
  int lane = threadIdx.x & 31;
  int n  = lane & 15;
  int kb = (lane >> 4) ? 8 : 0;
  const _Float16* p = W + (size_t)n * ldw + kb;
  v16h b;
#pragma unroll
  for (int g = 0; g < 8; ++g) { int k = (g < 4) ? (2*g) : (8 + 2*g); b[2*g] = p[k]; b[2*g+1] = p[k+1]; }
  return b;
}

static __device__ __forceinline__ v8f wmma_f16(v16h a, v16h b, v8f c) {
  return __builtin_amdgcn_wmma_f32_16x16x32_f16(false, a, false, b, (short)0, c, false, false);
}

// CDNA5 async global->LDS copy (ASYNCcnt-tracked), 16 bytes per lane.
static __device__ __forceinline__ void async_copy_b128(const void* gptr, void* lptr) {
  unsigned lds_off = (unsigned)(size_t)lptr;               // low 32 bits = LDS offset
  unsigned long long ga = (unsigned long long)(size_t)gptr;
  asm volatile("global_load_async_to_lds_b128 %0, %1, off"
               :: "v"(lds_off), "v"(ga) : "memory");
}
static __device__ __forceinline__ void wait_async0() {
  asm volatile("s_wait_asynccnt 0x0" ::: "memory");
}

// ---------------- Kernels ----------------

__global__ void cvt_f32_f16(const float* __restrict__ in, _Float16* __restrict__ out, int n) {
  int i = blockIdx.x * blockDim.x + threadIdx.x;
  if (i < n) out[i] = (_Float16)in[i];
}

// QKV projection p = xh @ wh^T + bias, scattering q/k/v into attention layouts.
// grid = (N_/64, D3/64), block = 128 (4 waves). Block computes a 64x64 tile.
// Double-buffered LDS staging: async copies of panel k+1 overlap WMMAs of panel k.
// pass 0 (row attn):  q,k -> [H][L][S][dh],  v -> [H][S][dh][L]  (transposed)
// pass 1 (col attn):  q,k -> [H][L][S][dh],  v -> [H][L][dh][S]  (transposed)
__global__ void qkv_gemm(const _Float16* __restrict__ xh,
                         const _Float16* __restrict__ wh,
                         const float* __restrict__ bias,
                         _Float16* __restrict__ q,
                         _Float16* __restrict__ k,
                         _Float16* __restrict__ v,
                         int pass) {
  __shared__ __align__(16) _Float16 lA[2][64 * 32];
  __shared__ __align__(16) _Float16 lB[2][64 * 32];
  const int m0 = blockIdx.x * 64;
  const int n0 = blockIdx.y * 64;
  const int tid  = threadIdx.x;       // 0..127
  const int wave = tid >> 5;          // 0..3

  // Per-thread copy coordinates: 16B chunk (row, colh) and (row+32, colh).
  const int crow  = tid >> 2;         // 0..31
  const int ccolh = (tid & 3) * 8;    // 0,8,16,24 halfs

  auto stage = [&](int buf, int k0) {
#pragma unroll
    for (int cc = 0; cc < 2; ++cc) {
      int row = crow + cc * 32;
      async_copy_b128(xh + (size_t)(m0 + row) * D_ + k0 + ccolh, &lA[buf][row * 32 + ccolh]);
      async_copy_b128(wh + (size_t)(n0 + row) * D_ + k0 + ccolh, &lB[buf][row * 32 + ccolh]);
    }
  };

  v8f acc[4] = {};
  const int NSTEP = D_ / 32;          // 24
  stage(0, 0);
  int buf = 0;
  for (int step = 0; step < NSTEP; ++step) {
    wait_async0();                    // my copies for 'step' have landed
    __syncthreads();                  // everyone's copies landed; prior compute done
    if (step + 1 < NSTEP) stage(buf ^ 1, (step + 1) * 32);   // overlap with WMMAs
    v16h a = load_a16(&lA[buf][wave * 16 * 32], 32);
#pragma unroll
    for (int t = 0; t < 4; ++t) {
      v16h b = load_b16_t(&lB[buf][t * 16 * 32], 32);
      acc[t] = wmma_f16(a, b, acc[t]);
    }
    buf ^= 1;
  }

  // Epilogue: part/h are block-uniform (64-col tiles never straddle 768 boundary).
  const int part = n0 / D_;            // 0=q, 1=k, 2=v (uniform)
  const int w0   = n0 - part * D_;     // multiple of 64
  const int h    = w0 / DH;            // uniform head index
  const int lane = tid & 31;
  const int nlo  = lane & 15;
  const int mhi  = (lane >> 4) ? 8 : 0;
  const int mbase = m0 + wave * 16;
#pragma unroll
  for (int t = 0; t < 4; ++t) {
    int c = t*16 + nlo;
    float bv = bias[n0 + t*16 + nlo];
#pragma unroll
    for (int r = 0; r < 8; ++r) {
      int m = mbase + r + mhi;
      int s = m >> 8;                  // m / L_
      int l = m & (L_ - 1);            // m % L_
      _Float16 hv = (_Float16)(acc[t][r] + bv);
      if (part == 0) {
        q[(((size_t)(h*L_ + l))*S_ + s)*DH + c] = hv;
      } else if (part == 1) {
        k[(((size_t)(h*L_ + l))*S_ + s)*DH + c] = hv;
      } else {
        if (pass == 0) v[(((size_t)(h*S_ + s))*DH + c)*L_ + l] = hv;   // [H][S][dh][L]
        else           v[(((size_t)(h*L_ + l))*DH + c)*S_ + s] = hv;   // [H][L][dh][S]
      }
    }
  }
}

// Tied-row logits: logits[h][i][j] = sum_{s,c} q[h][i][s*64+c] * k[h][j][s*64+c]
// grid = (L/16, L/16, H), block = 32.
__global__ void row_logits(const _Float16* __restrict__ q,
                           const _Float16* __restrict__ kk,
                           float* __restrict__ logits) {
  int i0 = blockIdx.x * 16, j0 = blockIdx.y * 16, h = blockIdx.z;
  const int KD = S_ * DH;  // 8192
  const _Float16* A = q  + ((size_t)(h*L_ + i0)) * KD;
  const _Float16* B = kk + ((size_t)(h*L_ + j0)) * KD;
  v8f acc = {};
  for (int k0 = 0; k0 < KD; k0 += 32) {
    v16h a = load_a16(A + k0, KD);
    v16h b = load_b16_t(B + k0, KD);
    acc = wmma_f16(a, b, acc);
  }
  int lane = threadIdx.x & 31;
  int n  = j0 + (lane & 15);
  int mb = i0 + ((lane >> 4) ? 8 : 0);
  float* out = logits + (size_t)h * L_ * L_;
#pragma unroll
  for (int r = 0; r < 8; ++r) out[(size_t)(mb + r) * L_ + n] = acc[r];
}

// Column logits: logits[h][l][i][j] = sum_c q[h][l][i][c] * k[h][l][j][c]
// grid = (S/16, L, H), block = 32; each wave computes a 16x128 row-tile.
__global__ void col_logits(const _Float16* __restrict__ q,
                           const _Float16* __restrict__ kk,
                           float* __restrict__ logits) {
  int h = blockIdx.z, l = blockIdx.y, i0 = blockIdx.x * 16;
  const _Float16* Ab = q  + ((size_t)(h*L_ + l)*S_ + i0) * DH;
  const _Float16* Bb = kk + ((size_t)(h*L_ + l)*S_) * DH;
  v8f acc[8] = {};
  for (int k0 = 0; k0 < DH; k0 += 32) {
    v16h a = load_a16(Ab + k0, DH);
#pragma unroll
    for (int t = 0; t < 8; ++t) {
      v16h b = load_b16_t(Bb + (size_t)(t*16) * DH + k0, DH);
      acc[t] = wmma_f16(a, b, acc[t]);
    }
  }
  int lane = threadIdx.x & 31;
  int nlo = lane & 15, mhi = (lane >> 4) ? 8 : 0;
  float* out = logits + ((size_t)(h*L_ + l)*S_) * S_;
#pragma unroll
  for (int t = 0; t < 8; ++t)
#pragma unroll
    for (int r = 0; r < 8; ++r)
      out[(size_t)(i0 + r + mhi) * S_ + (t*16 + nlo)] = acc[t][r];
}

// In-place softmax over last dim. grid = rows, block = 32 (one wave per row).
__global__ void softmax_rows(float* __restrict__ x, int C) {
  float* p = x + (size_t)blockIdx.x * C;
  int lane = threadIdx.x;
  float mx = -3.402823466e38f;
  for (int c = lane; c < C; c += 32) mx = fmaxf(mx, p[c]);
#pragma unroll
  for (int o = 16; o > 0; o >>= 1) mx = fmaxf(mx, __shfl_xor(mx, o, 32));
  float sum = 0.f;
  for (int c = lane; c < C; c += 32) { float e = __expf(p[c] - mx); p[c] = e; sum += e; }
#pragma unroll
  for (int o = 16; o > 0; o >>= 1) sum += __shfl_xor(sum, o, 32);
  float inv = 1.0f / sum;
  for (int c = lane; c < C; c += 32) p[c] *= inv;
}

// attn @ V with V stored transposed ([..][d][j]) so B fragments are contiguous.
// colpass=0: out[s*L+i][h*64+d] = sum_j probs[h][i][j]    * vt[h][s][d][j]
// colpass=1: out[i*L+l][h*64+d] = sum_j probs[h][l][i][j] * vt[h][l][d][j]
// grid = (i_tiles, inner, H), block = 32; wave computes 16 x 64 (full dh).
__global__ void attn_av(const float* __restrict__ probs,
                        const _Float16* __restrict__ vt,
                        float* __restrict__ y, int colpass) {
  int h = blockIdx.z;
  int inner = blockIdx.y;          // s (row pass) or l (col pass)
  int i0 = blockIdx.x * 16;
  const float* A;
  const _Float16* B;
  int K, lda, ldb;
  if (!colpass) {
    K = L_; lda = L_; ldb = L_;
    A = probs + (size_t)h * L_ * L_ + (size_t)i0 * L_;
    B = vt + ((size_t)(h*S_ + inner) * DH) * L_;
  } else {
    K = S_; lda = S_; ldb = S_;
    A = probs + ((size_t)(h*L_ + inner)*S_ + i0) * S_;
    B = vt + ((size_t)(h*L_ + inner) * DH) * S_;
  }
  v8f acc[4] = {};
  for (int k0 = 0; k0 < K; k0 += 32) {
    v16h a = load_a16_f32(A + k0, lda);
#pragma unroll
    for (int t = 0; t < 4; ++t) {
      v16h b = load_b16_t(B + (size_t)(t*16) * ldb + k0, ldb);
      acc[t] = wmma_f16(a, b, acc[t]);
    }
  }
  int lane = threadIdx.x & 31;
  int nlo = lane & 15, mhi = (lane >> 4) ? 8 : 0;
#pragma unroll
  for (int t = 0; t < 4; ++t)
#pragma unroll
    for (int r = 0; r < 8; ++r) {
      int i = i0 + r + mhi;
      int d = t*16 + nlo;
      int m = colpass ? (i*L_ + inner) : (inner*L_ + i);
      y[(size_t)m * D_ + h*DH + d] = acc[t][r];
    }
}

// out = LayerNorm(resid + y) * g + beta; writes f32 and optional f16 copy.
// grid = N_, block = 256 (3 elements / thread).
__global__ void add_ln(const float* __restrict__ resid,
                       const float* __restrict__ y,
                       const float* __restrict__ g,
                       const float* __restrict__ beta,
                       float* __restrict__ out32,
                       _Float16* __restrict__ out16) {
  __shared__ float red[256];
  int row = blockIdx.x;
  const float* rp = resid + (size_t)row * D_;
  const float* yp = y     + (size_t)row * D_;
  float vals[3];
  float lsum = 0.f;
#pragma unroll
  for (int e = 0; e < 3; ++e) {
    int c = threadIdx.x + e*256;
    vals[e] = rp[c] + yp[c];
    lsum += vals[e];
  }
  red[threadIdx.x] = lsum; __syncthreads();
  for (int s = 128; s > 0; s >>= 1) {
    if ((int)threadIdx.x < s) red[threadIdx.x] += red[threadIdx.x + s];
    __syncthreads();
  }
  float mean = red[0] * (1.0f / D_);
  __syncthreads();
  float lvar = 0.f;
#pragma unroll
  for (int e = 0; e < 3; ++e) { float d = vals[e] - mean; lvar += d*d; }
  red[threadIdx.x] = lvar; __syncthreads();
  for (int s = 128; s > 0; s >>= 1) {
    if ((int)threadIdx.x < s) red[threadIdx.x] += red[threadIdx.x + s];
    __syncthreads();
  }
  float rstd = rsqrtf(red[0] * (1.0f / D_) + 1e-5f);
#pragma unroll
  for (int e = 0; e < 3; ++e) {
    int c = threadIdx.x + e*256;
    float o = (vals[e] - mean) * rstd * g[c] + beta[c];
    out32[(size_t)row * D_ + c] = o;
    if (out16) out16[(size_t)row * D_ + c] = (_Float16)o;
  }
}

// ---------------- Host launch ----------------

extern "C" void kernel_launch(void* const* d_in, const int* in_sizes, int n_in,
                              void* d_out, int out_size, void* d_ws, size_t ws_size,
                              hipStream_t stream) {
  (void)in_sizes; (void)n_in; (void)out_size; (void)ws_size;

  const float* x     = (const float*)d_in[0];
  const float* w_row = (const float*)d_in[1];
  const float* b_row = (const float*)d_in[2];
  const float* w_col = (const float*)d_in[3];
  const float* b_col = (const float*)d_in[4];
  const float* g1    = (const float*)d_in[5];
  const float* be1   = (const float*)d_in[6];
  const float* g2    = (const float*)d_in[7];
  const float* be2   = (const float*)d_in[8];
  float* out = (float*)d_out;

  // Workspace carve-out (all sizes multiples of 256 bytes).
  char* ws = (char*)d_ws;
  size_t off = 0;
  const size_t SZ_XH  = (size_t)N_ * D_ * 2;           //  50.3 MB f16 activations
  const size_t SZ_WH  = (size_t)D3 * D_ * 2;           //   3.5 MB f16 weights
  const size_t SZ_QKV = (size_t)H_ * L_ * S_ * DH * 2; //  50.3 MB each
  const size_t SZ_LOG = (size_t)H_ * L_ * S_ * S_ * 4; // 201.3 MB (covers row 3 MB too)
  const size_t SZ_Y   = (size_t)N_ * D_ * 4;           // 100.7 MB
  _Float16* xh   = (_Float16*)(ws + off); off += SZ_XH;
  _Float16* wh   = (_Float16*)(ws + off); off += SZ_WH;
  _Float16* qb   = (_Float16*)(ws + off); off += SZ_QKV;
  _Float16* kb   = (_Float16*)(ws + off); off += SZ_QKV;
  _Float16* vb   = (_Float16*)(ws + off); off += SZ_QKV;
  float*    logi = (float*)   (ws + off); off += SZ_LOG;
  float*    yb   = (float*)   (ws + off); off += SZ_Y;
  float*    out1 = (float*)   (ws + off); off += SZ_Y;

  // ---- Pass 1: tied-row attention ----
  {
    int n = N_ * D_;
    cvt_f32_f16<<<(n + 255)/256, 256, 0, stream>>>(x, xh, n);
  }
  {
    int n = D3 * D_;
    cvt_f32_f16<<<(n + 255)/256, 256, 0, stream>>>(w_row, wh, n);
  }
  qkv_gemm<<<dim3(N_/64, D3/64), 128, 0, stream>>>(xh, wh, b_row, qb, kb, vb, 0);
  row_logits<<<dim3(L_/16, L_/16, H_), 32, 0, stream>>>(qb, kb, logi);
  softmax_rows<<<H_ * L_, 32, 0, stream>>>(logi, L_);
  attn_av<<<dim3(L_/16, S_, H_), 32, 0, stream>>>(logi, vb, yb, 0);
  add_ln<<<N_, 256, 0, stream>>>(x, yb, g1, be1, out1, xh);   // xh <- f16(out1)

  // ---- Pass 2: column attention ----
  {
    int n = D3 * D_;
    cvt_f32_f16<<<(n + 255)/256, 256, 0, stream>>>(w_col, wh, n);
  }
  qkv_gemm<<<dim3(N_/64, D3/64), 128, 0, stream>>>(xh, wh, b_col, qb, kb, vb, 1);
  col_logits<<<dim3(S_/16, L_, H_), 32, 0, stream>>>(qb, kb, logi);
  softmax_rows<<<H_ * L_ * S_, 32, 0, stream>>>(logi, S_);
  attn_av<<<dim3(S_/16, L_, H_), 32, 0, stream>>>(logi, vb, yb, 1);
  add_ln<<<N_, 256, 0, stream>>>(out1, yb, g2, be2, out, nullptr);
}
